// FasterRCNN_15633680957859
// MI455X (gfx1250) — matrix operand order repaired and checked by reading, not compiled, not run on hardware
//
#include <hip/hip_runtime.h>

// ---------------------------------------------------------------- constants
#define IMGM1   511.0f      // IMG_W-1 == IMG_H-1
#define FDIM    32          // feature map H=W
#define NPOS    1024        // 32*32
#define CIN     3
#define CFEAT   128
#define NA      9           // anchors per position
#define NANCH   9216        // 1024*9
#define KPROP   2000
#define HID     1024
#define KBB     768         // 3*16*16
#define KRPN    1152        // 128*3*3
#define KFC1    6272        // 128*7*7
#define NPAD1   16384
#define NPAD2   2048

typedef __attribute__((ext_vector_type(16))) __bf16 v16bf;
typedef __attribute__((ext_vector_type(8)))  float  v8f;

static __device__ __forceinline__ unsigned short f32_to_bf16(float f) {
    unsigned int u = __float_as_uint(f);
    u += 0x7FFFu + ((u >> 16) & 1u);          // round-to-nearest-even
    return (unsigned short)(u >> 16);
}
static __device__ __forceinline__ int imin(int a, int b) { return a < b ? a : b; }
static __device__ __forceinline__ int imax(int a, int b) { return a > b ? a : b; }

// ---------------------------------------------------------------- helpers
__global__ void cvt_bf16_kernel(const float* __restrict__ src,
                                unsigned short* __restrict__ dst, int n) {
    int i = blockIdx.x * blockDim.x + threadIdx.x;
    if (i < n) dst[i] = f32_to_bf16(src[i]);
}

// im2col for backbone conv: 16x16 stride-16 VALID -> (1024 x 768) bf16
__global__ void im2col_bb_kernel(const float* __restrict__ x,
                                 unsigned short* __restrict__ dst) {
    int t = blockIdx.x * blockDim.x + threadIdx.x;
    if (t >= NPOS * KBB) return;
    int k = t % KBB, m = t / KBB;
    int ox = m % FDIM, oy = m / FDIM;
    int j = k % 16, rest = k / 16;
    int i = rest % 16, c = rest / 16;
    float v = x[(c * 512 + oy * 16 + i) * 512 + ox * 16 + j];
    dst[t] = f32_to_bf16(v);
}

// im2col for RPN 3x3 SAME conv over feat[(y*32+x)*128+c] -> (1024 x 1152) bf16
__global__ void im2col_rpn_kernel(const float* __restrict__ feat,
                                  unsigned short* __restrict__ dst) {
    int t = blockIdx.x * blockDim.x + threadIdx.x;
    if (t >= NPOS * KRPN) return;
    int k = t % KRPN, m = t / KRPN;
    int x = m % FDIM, y = m / FDIM;
    int dx = k % 3, rest = k / 3;
    int dy = rest % 3, c = rest / 3;
    int yy = y + dy - 1, xx = x + dx - 1;
    float v = 0.f;
    if (yy >= 0 && yy < FDIM && xx >= 0 && xx < FDIM)
        v = feat[(yy * FDIM + xx) * CFEAT + c];
    dst[t] = f32_to_bf16(v);
}

// ---------------------------------------------------------------- WMMA GEMM
// out[m,n] = relu( sum_k A[m,k] * W[n,k] + bias[n] )
// A: (M x K) bf16 row-major, W: (N x K) bf16 row-major.
// 2x2 register macro-tile (32x32 of D) per wave: each A/B fragment feeds 2 WMMAs.
__global__ __launch_bounds__(256)
void gemm_bf16_wmma(const unsigned short* __restrict__ A,
                    const unsigned short* __restrict__ W,
                    const float* __restrict__ bias,
                    float* __restrict__ outF,
                    unsigned short* __restrict__ outB,
                    int M, int N, int K, int relu) {
    const int wave = threadIdx.x >> 5;
    const int lane = threadIdx.x & 31;
    const int tilesM = M >> 4, tilesN = N >> 4;
    const int mtM = (tilesM + 1) >> 1;            // macro rows (may have odd tail)
    const int mtN = tilesN >> 1;                  // N tiles always even here
    int mt = blockIdx.x * 8 + wave;
    if (mt >= mtM * mtN) return;                  // wave-uniform: EXEC stays all-1
    const int m0 = (mt / mtN) * 2;
    const int n0 = (mt % mtN) * 2;
    const bool hasM1 = (m0 + 1) < tilesM;         // wave-uniform tail guard
    const int hf = lane >> 4;                     // lane half
    const int r  = lane & 15;                     // A row / B col / D col

    const unsigned short* a0 = A + (size_t)(m0 * 16 + r) * K;
    const unsigned short* a1 = A + (size_t)((hasM1 ? m0 + 1 : m0) * 16 + r) * K;
    const unsigned short* w0 = W + (size_t)(n0 * 16 + r) * K;
    const unsigned short* w1 = W + (size_t)((n0 + 1) * 16 + r) * K;

    // A 16x32 bf16 layout: VGPR j<4: K = hf*8 + 2j ; j>=4: K = 16 + hf*8 + 2(j-4)
    auto loadA = [&](const unsigned short* row, int k0) {
        union { v16bf v; unsigned int u[8]; } f;
#pragma unroll
        for (int j = 0; j < 8; ++j) {
            int ka = ((j >= 4) ? 16 : 0) + hf * 8 + (j & 3) * 2;
            f.u[j] = *(const unsigned int*)(row + k0 + ka);
        }
        return f.v;
    };
    // B 32x16 bf16 layout: same K order as A (VGPR j<4: K = hf*8 + 2j ; j>=4: K = 16 + hf*8 + 2(j-4))
    auto loadB = [&](const unsigned short* row, int k0) {
        union { v16bf v; unsigned int u[8]; } f;
#pragma unroll
        for (int j = 0; j < 8; ++j)
            f.u[j] = *(const unsigned int*)(row + k0 + ((j >= 4) ? 16 : 0) + hf * 8 + (j & 3) * 2);
        return f.v;
    };

    v8f acc00 = {}, acc01 = {}, acc10 = {}, acc11 = {};
    for (int k0 = 0; k0 < K; k0 += 32) {
        v16bf fa0 = loadA(a0, k0);
        v16bf fa1 = loadA(a1, k0);
        v16bf fb0 = loadB(w0, k0);
        v16bf fb1 = loadB(w1, k0);
        if (k0 + 32 < K) {
            __builtin_prefetch(a0 + k0 + 32, 0, 3);
            __builtin_prefetch(a1 + k0 + 32, 0, 3);
            __builtin_prefetch(w0 + k0 + 32, 0, 3);
            __builtin_prefetch(w1 + k0 + 32, 0, 3);
        }
        acc00 = __builtin_amdgcn_wmma_f32_16x16x32_bf16(false, fa0, false, fb0, (short)0, acc00, false, false);
        acc01 = __builtin_amdgcn_wmma_f32_16x16x32_bf16(false, fa0, false, fb1, (short)0, acc01, false, false);
        acc10 = __builtin_amdgcn_wmma_f32_16x16x32_bf16(false, fa1, false, fb0, (short)0, acc10, false, false);
        acc11 = __builtin_amdgcn_wmma_f32_16x16x32_bf16(false, fa1, false, fb1, (short)0, acc11, false, false);
    }

    auto store_tile = [&](v8f acc, int tm, int tn) {
        float bv = bias ? bias[tn * 16 + r] : 0.f;
#pragma unroll
        for (int q = 0; q < 8; ++q) {
            int m = tm * 16 + q + hf * 8;          // D layout: VGPR q -> row q + hf*8
            int n = tn * 16 + r;
            float v = acc[q] + bv;
            if (relu) v = fmaxf(v, 0.f);
            size_t o = (size_t)m * N + n;
            if (outF) outF[o] = v;
            if (outB) outB[o] = f32_to_bf16(v);
        }
    };
    store_tile(acc00, m0, n0);
    store_tile(acc01, m0, n0 + 1);
    if (hasM1) {
        store_tile(acc10, m0 + 1, n0);
        store_tile(acc11, m0 + 1, n0 + 1);
    }
}

// ---------------------------------------------------------------- small heads
// out[m*N+n] = X[m,:K] . W[n,:K] + b[n]   (fp32, N tiny: 18/36/2/4)
__global__ void dense_rows_kernel(const float* __restrict__ X, int M, int K,
                                  const float* __restrict__ W,
                                  const float* __restrict__ b,
                                  float* __restrict__ out, int N) {
    int t = blockIdx.x * blockDim.x + threadIdx.x;
    if (t >= M * N) return;
    int m = t / N, n = t % N;
    const float* xp = X + (size_t)m * K;
    const float* wp = W + (size_t)n * K;
    float acc = 0.f;
    for (int k = 0; k < K; ++k) acc = fmaf(xp[k], wp[k], acc);
    out[t] = acc + (b ? b[n] : 0.f);
}

__global__ void anchors_kernel(float* __restrict__ anch) {
    int i = blockIdx.x * blockDim.x + threadIdx.x;
    if (i >= NANCH) return;
    const float S[3] = {32.f, 64.f, 128.f};
    const float R[3] = {0.5f, 1.f, 2.f};
    int a = i % NA, pos = i / NA;
    int x = pos % FDIM, y = pos / FDIM;
    float sq = sqrtf(R[a % 3]);
    float wv = S[a / 3] / sq, hv = S[a / 3] * sq;
    float cx = (x + 0.5f) * 16.f, cy = (y + 0.5f) * 16.f;
    anch[i * 4 + 0] = cx - 0.5f * wv;
    anch[i * 4 + 1] = cy - 0.5f * hv;
    anch[i * 4 + 2] = cx + 0.5f * wv;
    anch[i * 4 + 3] = cy + 0.5f * hv;
}

__global__ void softmax2_kernel(const float* __restrict__ lg,
                                float* __restrict__ sc, int n) {
    int i = blockIdx.x * blockDim.x + threadIdx.x;
    if (i >= n) return;
    float l0 = lg[2 * i], l1 = lg[2 * i + 1];
    float m = fmaxf(l0, l1);
    float e0 = __expf(l0 - m), e1 = __expf(l1 - m);
    sc[i] = e1 / (e0 + e1);
}

__global__ void decode_clip_kernel(const float* __restrict__ del,
                                   const float* __restrict__ anc,
                                   float* __restrict__ out, int n) {
    int i = blockIdx.x * blockDim.x + threadIdx.x;
    if (i >= n) return;
    float a0 = anc[i*4], a1 = anc[i*4+1], a2 = anc[i*4+2], a3 = anc[i*4+3];
    float w = a2 - a0 + 1.f, h = a3 - a1 + 1.f;
    float cx = a0 + 0.5f * w, cy = a1 + 0.5f * h;
    float dx = del[i*4], dy = del[i*4+1], dw = del[i*4+2], dh = del[i*4+3];
    float pcx = dx * w + cx, pcy = dy * h + cy;
    float pw = __expf(dw) * w, ph = __expf(dh) * h;
    out[i*4+0] = fminf(fmaxf(pcx - 0.5f * pw, 0.f), IMGM1);
    out[i*4+1] = fminf(fmaxf(pcy - 0.5f * ph, 0.f), IMGM1);
    out[i*4+2] = fminf(fmaxf(pcx + 0.5f * pw, 0.f), IMGM1);
    out[i*4+3] = fminf(fmaxf(pcy + 0.5f * ph, 0.f), IMGM1);
}

// ---------------------------------------------------------------- sort + NMS
__global__ void sort_prep_kernel(const float* __restrict__ sc,
                                 float* __restrict__ key, int* __restrict__ idx,
                                 int n, int npad) {
    int i = blockIdx.x * blockDim.x + threadIdx.x;
    if (i >= npad) return;
    key[i] = (i < n) ? -sc[i] : 3.4e38f;   // ascending sort of -score == descending
    idx[i] = (i < n) ? i : 0;
}

__global__ void bitonic_step_kernel(float* __restrict__ key, int* __restrict__ idx,
                                    int n, int k, int j) {
    int i = blockIdx.x * blockDim.x + threadIdx.x;
    if (i >= n) return;
    int ixj = i ^ j;
    if (ixj > i) {
        bool asc = ((i & k) == 0);
        float ki = key[i], kj = key[ixj];
        if (asc ? (ki > kj) : (ki < kj)) {
            key[i] = kj; key[ixj] = ki;
            int t = idx[i]; idx[i] = idx[ixj]; idx[ixj] = t;
        }
    }
}

__global__ void gather4_kernel(const float* __restrict__ src,
                               const int* __restrict__ ord,
                               float* __restrict__ dst, int n) {
    int i = blockIdx.x * blockDim.x + threadIdx.x;
    if (i >= n) return;
    int s = ord[i];
    dst[i*4+0] = src[s*4+0]; dst[i*4+1] = src[s*4+1];
    dst[i*4+2] = src[s*4+2]; dst[i*4+3] = src[s*4+3];
}

// greedy NMS on score-sorted boxes; single workgroup, supp vector in LDS
__global__ __launch_bounds__(1024)
void nms_kernel(const float* __restrict__ sb, const int* __restrict__ order,
                int N, float thr, int K, int* __restrict__ keep,
                int* __restrict__ valid) {
    __shared__ unsigned char supp[NANCH];
    for (int i = threadIdx.x; i < N; i += blockDim.x) supp[i] = 0;
    __syncthreads();
    for (int i = 0; i < N - 1; ++i) {
        if (!supp[i]) {
            float x1 = sb[i*4+0], y1 = sb[i*4+1], x2 = sb[i*4+2], y2 = sb[i*4+3];
            float ai = (x2 - x1 + 1.f) * (y2 - y1 + 1.f);
            for (int j = i + 1 + threadIdx.x; j < N; j += blockDim.x) {
                if (supp[j]) continue;
                float bx1 = sb[j*4+0], by1 = sb[j*4+1], bx2 = sb[j*4+2], by2 = sb[j*4+3];
                float xx1 = fmaxf(x1, bx1), yy1 = fmaxf(y1, by1);
                float xx2 = fminf(x2, bx2), yy2 = fminf(y2, by2);
                float inter = fmaxf(xx2 - xx1 + 1.f, 0.f) * fmaxf(yy2 - yy1 + 1.f, 0.f);
                float aj = (bx2 - bx1 + 1.f) * (by2 - by1 + 1.f);
                if (inter / (ai + aj - inter) > thr) supp[j] = 1;
            }
        }
        __syncthreads();
    }
    if (threadIdx.x == 0) {
        for (int k = 0; k < K; ++k) keep[k] = 0;
        int rank = 0;
        for (int i = 0; i < N; ++i)
            if (!supp[i]) { if (rank < K) keep[rank] = order[i]; ++rank; }
        int tot = imin(rank, K);
        for (int k = 0; k < K; ++k) valid[k] = (k < tot) ? 1 : 0;
    }
}

// ---------------------------------------------------------------- crop+pool
// bilinear 14x14 crop of feat[(y*32+x)*128+c] then 2x2 max -> bf16 (KPROP x 6272)
__global__ void crop_pool_kernel(const float* __restrict__ feat,
                                 const float* __restrict__ props,
                                 unsigned short* __restrict__ out) {
    int t = blockIdx.x * blockDim.x + threadIdx.x;
    if (t >= KPROP * CFEAT * 49) return;
    int ox = t % 7, tmp = t / 7;
    int oy = tmp % 7; tmp /= 7;
    int c = tmp % CFEAT, p = tmp / CFEAT;
    float nx1 = props[p*4+0] / IMGM1, ny1 = props[p*4+1] / IMGM1;
    float nx2 = props[p*4+2] / IMGM1, ny2 = props[p*4+3] / IMGM1;
    float mv = -3.4e38f;
#pragma unroll
    for (int sy = 0; sy < 2; ++sy)
#pragma unroll
        for (int sx = 0; sx < 2; ++sx) {
            float ty = (float)(oy * 2 + sy) / 13.f;
            float tx = (float)(ox * 2 + sx) / 13.f;
            float ys = (ny1 + ty * (ny2 - ny1)) * 31.f;
            float xs = (nx1 + tx * (nx2 - nx1)) * 31.f;
            int y0 = imin(imax((int)floorf(ys), 0), 31);
            int x0 = imin(imax((int)floorf(xs), 0), 31);
            int y1i = imin(y0 + 1, 31), x1i = imin(x0 + 1, 31);
            float wy = ys - (float)y0, wx = xs - (float)x0;
            float f00 = feat[(y0 * 32 + x0) * CFEAT + c];
            float f01 = feat[(y0 * 32 + x1i) * CFEAT + c];
            float f10 = feat[(y1i * 32 + x0) * CFEAT + c];
            float f11 = feat[(y1i * 32 + x1i) * CFEAT + c];
            float top = f00 * (1.f - wx) + f01 * wx;
            float bot = f10 * (1.f - wx) + f11 * wx;
            mv = fmaxf(mv, top * (1.f - wy) + bot * wy);
        }
    out[(size_t)p * KFC1 + c * 49 + oy * 7 + ox] = f32_to_bf16(mv);
}

__global__ void finalize_kernel(const float* __restrict__ dets,
                                const float* __restrict__ rsc,
                                const int* __restrict__ keep,
                                const int* __restrict__ valid,
                                float* __restrict__ odets,
                                float* __restrict__ osc) {
    int k = blockIdx.x * blockDim.x + threadIdx.x;
    if (k >= KPROP) return;
    int s = keep[k];
    float m = valid[k] ? 1.f : 0.f;
    odets[k*4+0] = dets[s*4+0] * m; odets[k*4+1] = dets[s*4+1] * m;
    odets[k*4+2] = dets[s*4+2] * m; odets[k*4+3] = dets[s*4+3] * m;
    osc[k] = rsc[s] * m;
}

// ---------------------------------------------------------------- launch
static inline unsigned gb(size_t n, unsigned b) { return (unsigned)((n + b - 1) / b); }
static inline unsigned gemm_grid(int M, int N) {
    int tm = M / 16, tn = N / 16;
    int mt = ((tm + 1) / 2) * (tn / 2);
    return (unsigned)((mt + 7) / 8);
}

extern "C" void kernel_launch(void* const* d_in, const int* in_sizes, int n_in,
                              void* d_out, int out_size, void* d_ws, size_t ws_size,
                              hipStream_t stream) {
    (void)in_sizes; (void)n_in; (void)out_size; (void)ws_size;
    const float* x         = (const float*)d_in[0];
    const float* bb_w      = (const float*)d_in[1];
    const float* bb_b      = (const float*)d_in[2];
    const float* rpn_w     = (const float*)d_in[3];
    const float* rpn_b     = (const float*)d_in[4];
    const float* rpn_cls_w = (const float*)d_in[5];
    const float* rpn_cls_b = (const float*)d_in[6];
    const float* rpn_reg_w = (const float*)d_in[7];
    const float* rpn_reg_b = (const float*)d_in[8];
    const float* fc1_w     = (const float*)d_in[9];
    const float* fc1_b     = (const float*)d_in[10];
    const float* fc2_w     = (const float*)d_in[11];
    const float* fc2_b     = (const float*)d_in[12];
    const float* cls_w     = (const float*)d_in[13];
    const float* cls_b     = (const float*)d_in[14];
    const float* reg_w     = (const float*)d_in[15];
    const float* reg_b     = (const float*)d_in[16];
    float* out = (float*)d_out;

    // output regions (floats)
    float* o_logits = out;                   // 9216*2
    float* o_deltas = out + 18432;           // 9216*4
    float* o_props  = out + 55296;           // 2000*4
    float* o_anch   = out + 63296;           // 9216*4
    float* o_rlog   = out + 100160;          // 2000*2
    float* o_rdel   = out + 104160;          // 2000*4
    float* o_dets   = out + 112160;          // 2000*4
    float* o_osc    = out + 120160;          // 2000

    // workspace carve-up (256B aligned)
    char* base = (char*)d_ws;
    size_t off = 0;
    auto carve = [&](size_t bytes) {
        char* p = base + off;
        off = (off + bytes + 255) & ~(size_t)255;
        return p;
    };
    unsigned short* icbb  = (unsigned short*)carve((size_t)NPOS * KBB * 2);
    unsigned short* wbb   = (unsigned short*)carve((size_t)CFEAT * KBB * 2);
    float*          feat  = (float*)carve((size_t)NPOS * CFEAT * 4);
    unsigned short* icrpn = (unsigned short*)carve((size_t)NPOS * KRPN * 2);
    unsigned short* wrpn  = (unsigned short*)carve((size_t)CFEAT * KRPN * 2);
    float*          hbuf  = (float*)carve((size_t)NPOS * CFEAT * 4);
    float*          scbuf = (float*)carve((size_t)NANCH * 4);
    float*          boxes = (float*)carve((size_t)NANCH * 4 * 4);
    float*          skey  = (float*)carve((size_t)NPAD1 * 4);
    int*            sidx  = (int*)carve((size_t)NPAD1 * 4);
    float*          sbox  = (float*)carve((size_t)NANCH * 4 * 4);
    int*            keep1 = (int*)carve((size_t)KPROP * 4);
    int*            val1  = (int*)carve((size_t)KPROP * 4);
    unsigned short* crops = (unsigned short*)carve((size_t)KPROP * KFC1 * 2);
    unsigned short* wfc1  = (unsigned short*)carve((size_t)HID * KFC1 * 2);
    unsigned short* h1b   = (unsigned short*)carve((size_t)KPROP * HID * 2);
    unsigned short* wfc2  = (unsigned short*)carve((size_t)HID * HID * 2);
    float*          h2f   = (float*)carve((size_t)KPROP * HID * 4);
    float*          rsc   = (float*)carve((size_t)KPROP * 4);
    float*          dets  = (float*)carve((size_t)KPROP * 4 * 4);
    int*            keep2 = (int*)carve((size_t)KPROP * 4);
    int*            val2  = (int*)carve((size_t)KPROP * 4);

    const unsigned B = 256;

    // ---- backbone conv (im2col + WMMA GEMM, bias+relu) -> feat
    cvt_bf16_kernel<<<gb(CFEAT * KBB, B), B, 0, stream>>>(bb_w, wbb, CFEAT * KBB);
    im2col_bb_kernel<<<gb((size_t)NPOS * KBB, B), B, 0, stream>>>(x, icbb);
    gemm_bf16_wmma<<<gemm_grid(NPOS, CFEAT), B, 0, stream>>>(
        icbb, wbb, bb_b, feat, (unsigned short*)nullptr, NPOS, CFEAT, KBB, 1);

    // ---- RPN 3x3 conv -> hbuf
    cvt_bf16_kernel<<<gb(CFEAT * KRPN, B), B, 0, stream>>>(rpn_w, wrpn, CFEAT * KRPN);
    im2col_rpn_kernel<<<gb((size_t)NPOS * KRPN, B), B, 0, stream>>>(feat, icrpn);
    gemm_bf16_wmma<<<gemm_grid(NPOS, CFEAT), B, 0, stream>>>(
        icrpn, wrpn, rpn_b, hbuf, (unsigned short*)nullptr, NPOS, CFEAT, KRPN, 1);

    // ---- RPN heads (1x1 convs) straight into output regions
    dense_rows_kernel<<<gb(NPOS * 18, B), B, 0, stream>>>(hbuf, NPOS, CFEAT,
                                                          rpn_cls_w, rpn_cls_b, o_logits, 18);
    dense_rows_kernel<<<gb(NPOS * 36, B), B, 0, stream>>>(hbuf, NPOS, CFEAT,
                                                          rpn_reg_w, rpn_reg_b, o_deltas, 36);

    // ---- anchors, scores, decode
    anchors_kernel<<<gb(NANCH, B), B, 0, stream>>>(o_anch);
    softmax2_kernel<<<gb(NANCH, B), B, 0, stream>>>(o_logits, scbuf, NANCH);
    decode_clip_kernel<<<gb(NANCH, B), B, 0, stream>>>(o_deltas, o_anch, boxes, NANCH);

    // ---- sort by score (descending) + NMS #1 + gather proposals
    sort_prep_kernel<<<gb(NPAD1, B), B, 0, stream>>>(scbuf, skey, sidx, NANCH, NPAD1);
    for (int k = 2; k <= NPAD1; k <<= 1)
        for (int j = k >> 1; j > 0; j >>= 1)
            bitonic_step_kernel<<<NPAD1 / B, B, 0, stream>>>(skey, sidx, NPAD1, k, j);
    gather4_kernel<<<gb(NANCH, B), B, 0, stream>>>(boxes, sidx, sbox, NANCH);
    nms_kernel<<<1, 1024, 0, stream>>>(sbox, sidx, NANCH, 0.5f, KPROP, keep1, val1);
    gather4_kernel<<<gb(KPROP, B), B, 0, stream>>>(boxes, keep1, o_props, KPROP);

    // ---- ROI crop+bilinear+2x2 maxpool -> bf16 activations for FC1
    crop_pool_kernel<<<gb((size_t)KPROP * CFEAT * 49, B), B, 0, stream>>>(feat, o_props, crops);

    // ---- FC1 / FC2 (WMMA, bias+relu)
    cvt_bf16_kernel<<<gb(HID * KFC1, B), B, 0, stream>>>(fc1_w, wfc1, HID * KFC1);
    gemm_bf16_wmma<<<gemm_grid(KPROP, HID), B, 0, stream>>>(
        crops, wfc1, fc1_b, (float*)nullptr, h1b, KPROP, HID, KFC1, 1);
    cvt_bf16_kernel<<<gb(HID * HID, B), B, 0, stream>>>(fc2_w, wfc2, HID * HID);
    gemm_bf16_wmma<<<gemm_grid(KPROP, HID), B, 0, stream>>>(
        h1b, wfc2, fc2_b, h2f, (unsigned short*)nullptr, KPROP, HID, HID, 1);

    // ---- RCNN heads
    dense_rows_kernel<<<gb(KPROP * 2, B), B, 0, stream>>>(h2f, KPROP, HID, cls_w, cls_b, o_rlog, 2);
    dense_rows_kernel<<<gb(KPROP * 4, B), B, 0, stream>>>(h2f, KPROP, HID, reg_w, reg_b, o_rdel, 4);
    softmax2_kernel<<<gb(KPROP, B), B, 0, stream>>>(o_rlog, rsc, KPROP);
    decode_clip_kernel<<<gb(KPROP, B), B, 0, stream>>>(o_rdel, o_props, dets, KPROP);

    // ---- sort + NMS #2 + finalize
    sort_prep_kernel<<<gb(NPAD2, B), B, 0, stream>>>(rsc, skey, sidx, KPROP, NPAD2);
    for (int k = 2; k <= NPAD2; k <<= 1)
        for (int j = k >> 1; j > 0; j >>= 1)
            bitonic_step_kernel<<<NPAD2 / B, B, 0, stream>>>(skey, sidx, NPAD2, k, j);
    gather4_kernel<<<gb(KPROP, B), B, 0, stream>>>(dets, sidx, sbox, KPROP);
    nms_kernel<<<1, 1024, 0, stream>>>(sbox, sidx, KPROP, 0.3f, KPROP, keep2, val2);
    finalize_kernel<<<gb(KPROP, B), B, 0, stream>>>(dets, rsc, keep2, val2, o_dets, o_osc);
}
